// WhisperBiasing_55731495632965
// MI455X (gfx1250) — matrix operand order repaired and checked by reading, not compiled, not run on hardware
//
#include <hip/hip_runtime.h>
#include <hip/hip_bf16.h>

#define BB 2
#define TT 256
#define HH 768      // K dimension of both GEMMs
#define AA 512      // N dimension of both GEMMs
#define MM 128
#define VV 51864

typedef __attribute__((ext_vector_type(16))) __bf16 v16bf;
typedef __attribute__((ext_vector_type(8)))  float  v8f;

__device__ __forceinline__ void ldf8(const float* p, float* o) {
    float4 a = ((const float4*)p)[0];
    float4 b = ((const float4*)p)[1];
    o[0]=a.x; o[1]=a.y; o[2]=a.z; o[3]=a.w;
    o[4]=b.x; o[5]=b.y; o[6]=b.z; o[7]=b.w;
}

// C(rows x 512) = X(rows x 768) * W(512 x 768)^T + bias
// Row r >= mainrows reads from `lastrow` (the ooKB embedding) when lastrow != nullptr.
// Each wave computes a 64(M) x 16(N) strip: 4 WMMA accumulators, B fragment reused 4x.
__global__ __launch_bounds__(128)
void gemm_xwT_kernel(const float* __restrict__ X, const float* __restrict__ lastrow,
                     int rows, int mainrows,
                     const float* __restrict__ W, const float* __restrict__ bias,
                     float* __restrict__ C)
{
    const int lane = threadIdx.x & 31;
    const int wave = threadIdx.x >> 5;
    const int g  = lane >> 4;      // lane group (0: lanes 0-15, 1: lanes 16-31)
    const int ln = lane & 15;
    const int n0 = (blockIdx.y * 4 + wave) * 16;
    const int m0 = blockIdx.x * 64;

    // A-row pointers for the 4 M sub-tiles (A-matrix: M = lane%16 in both halves)
    const float* arow[4];
    #pragma unroll
    for (int s = 0; s < 4; ++s) {
        int r = m0 + s * 16 + ln;
        int rc = r < rows ? r : rows - 1;
        arow[s] = (lastrow != nullptr && rc >= mainrows) ? lastrow
                                                         : (X + (size_t)rc * HH);
    }
    const float* brow = W + (size_t)(n0 + ln) * HH;   // B col N = lane%16 is a row of W

    v8f acc[4];
    #pragma unroll
    for (int s = 0; s < 4; ++s)
        acc[s] = (v8f){0.f,0.f,0.f,0.f,0.f,0.f,0.f,0.f};

    for (int k0 = 0; k0 < HH; k0 += 32) {
        // B fragment (32x16 bf16): lane -> N=lane%16, element p -> K = 16*g + p
        float tb[16];
        ldf8(brow + k0 + 16 * g,     tb);
        ldf8(brow + k0 + 16 * g + 8, tb + 8);
        v16bf bf;
        #pragma unroll
        for (int i = 0; i < 16; ++i) bf[i] = (__bf16)tb[i];

        #pragma unroll
        for (int s = 0; s < 4; ++s) {
            // A fragment (16x32 bf16): element p -> K = 16*g + p
            float ta[16];
            ldf8(arow[s] + k0 + 16 * g,     ta);
            ldf8(arow[s] + k0 + 16 * g + 8, ta + 8);
            v16bf af;
            #pragma unroll
            for (int i = 0; i < 16; ++i) af[i] = (__bf16)ta[i];
            acc[s] = __builtin_amdgcn_wmma_f32_16x16x32_bf16(
                false, af, false, bf, (short)0, acc[s], false, false);
        }
    }

    const float bv = bias[n0 + ln];
    #pragma unroll
    for (int s = 0; s < 4; ++s) {
        #pragma unroll
        for (int r = 0; r < 8; ++r) {
            int row = m0 + s * 16 + r + 8 * g;   // C/D layout: M = vgpr + 8*(lane/16)
            if (row < rows)
                C[(size_t)row * AA + n0 + ln] = acc[s][r] + bv;
        }
    }
}

// Per-(b,t): scores over M, softmax, ooKB mass, Hptr (m<127), pg & scale.
__global__ __launch_bounds__(256)
void attend_kernel(const float* __restrict__ KBproj, const float* __restrict__ query,
                   const float* __restrict__ hidden, const int* __restrict__ index_list,
                   const int* __restrict__ step_mask, const int* __restrict__ p_gen_mask,
                   const float* __restrict__ Wg, const float* __restrict__ bg,
                   float* __restrict__ wts, float* __restrict__ pgs)
{
    __shared__ float q[AA];
    __shared__ float sc[MM];
    __shared__ int   idx[MM];
    __shared__ float red[256];
    __shared__ float red2[256];

    const int tid = threadIdx.x;
    const int bt  = blockIdx.x;                     // 0 .. B*T-1
    const float* qrow = query + (size_t)bt * AA;
    q[tid]       = qrow[tid];
    q[tid + 256] = qrow[tid + 256];
    if (tid < MM) idx[tid] = index_list[(size_t)bt * MM + tid];
    __syncthreads();

    // scores: 16 groups of 16 lanes, each group handles 8 m values
    const int gid = tid >> 4, ln = tid & 15;
    for (int m = gid; m < MM; m += 16) {
        const float* kr = KBproj + (size_t)idx[m] * AA;
        float acc = 0.f;
        for (int a = ln; a < AA; a += 16) acc += kr[a] * q[a];
        for (int off = 8; off > 0; off >>= 1) acc += __shfl_down(acc, off, 16);
        if (ln == 0) {
            float s = acc * 0.044194173824159216f;   // 1/sqrt(512)
            if (step_mask[(size_t)bt * MM + m]) s = -1.0e9f;
            sc[m] = s;
        }
    }
    __syncthreads();

    // softmax over sc[0..127]
    red[tid] = (tid < MM) ? sc[tid] : -1.0e30f;
    __syncthreads();
    for (int s = 128; s > 0; s >>= 1) {
        if (tid < s) red[tid] = fmaxf(red[tid], red[tid + s]);
        __syncthreads();
    }
    const float mx = red[0];
    __syncthreads();
    float e = 0.f;
    if (tid < MM) e = __expf(sc[tid] - mx);
    red[tid]  = e;
    red2[tid] = (tid < MM && idx[tid] == VV) ? e : 0.f;
    __syncthreads();
    for (int s = 128; s > 0; s >>= 1) {
        if (tid < s) { red[tid] += red[tid + s]; red2[tid] += red2[tid + s]; }
        __syncthreads();
    }
    const float denom = red[0];
    const float oob   = red2[0] / denom;
    __syncthreads();
    if (tid < MM) {
        float w = e / denom;
        sc[tid] = w;
        wts[(size_t)bt * MM + tid] = w;
    }
    __syncthreads();

    // Hptr components a = tid, tid+256 (exclude m = M-1)
    float h0 = 0.f, h1 = 0.f;
    for (int m = 0; m < MM - 1; ++m) {
        const float* kr = KBproj + (size_t)idx[m] * AA;
        float wm = sc[m];
        h0 += wm * kr[tid];
        h1 += wm * kr[tid + 256];
    }
    // gen = Wg[0:512].Hptr + Wg[512:1280].hidden + bg
    const float* hrow = hidden + (size_t)bt * HH;
    float part = Wg[tid] * h0 + Wg[tid + 256] * h1
               + Wg[AA + tid]       * hrow[tid]
               + Wg[AA + tid + 256] * hrow[tid + 256]
               + Wg[AA + tid + 512] * hrow[tid + 512];
    __syncthreads();
    red[tid] = part;
    __syncthreads();
    for (int s = 128; s > 0; s >>= 1) {
        if (tid < s) red[tid] += red[tid + s];
        __syncthreads();
    }
    if (tid == 0) {
        float gen = red[0] + bg[0];
        float pg  = 1.f / (1.f + __expf(-gen));
        if (p_gen_mask[bt]) pg = 0.f;
        pgs[bt * 2]     = pg;
        pgs[bt * 2 + 1] = 1.f - pg + pg * oob;   // (1 - pg + comp)
    }
}

// p_final row = softmax(logits[b,t,:]) * scale  (online softmax, 2 read passes)
__global__ __launch_bounds__(256)
void row_softmax_kernel(const float* __restrict__ logits, const float* __restrict__ pgs,
                        const int* __restrict__ sotp, float* __restrict__ pfin)
{
    __shared__ float rm[256], rs[256];
    const int sot  = *sotp;
    const int tlen = TT - sot;
    const int nrows = BB * tlen;
    const int r = blockIdx.x;
    if (r >= nrows) return;
    const int b = r / tlen, tt = r % tlen, t = tt + sot - 1;
    const float* lrow = logits + ((size_t)b * TT + t) * VV;
    const int tid = threadIdx.x;

    float mx = -1.0e30f, sm = 0.f;
    for (int v = tid; v < VV; v += 256) {
        float x = lrow[v];
        if (x > mx) { sm *= __expf(mx - x); mx = x; }
        sm += __expf(x - mx);
    }
    rm[tid] = mx; rs[tid] = sm;
    __syncthreads();
    if (tid == 0) {
        float M = rm[0], S = rs[0];
        for (int i = 1; i < 256; ++i) {
            float m2 = rm[i], s2 = rs[i];
            if (m2 > M) { S = S * __expf(M - m2) + s2; M = m2; }
            else        { S += s2 * __expf(m2 - M); }
        }
        rm[0] = M; rs[0] = S;
    }
    __syncthreads();
    const float M = rm[0];
    const float scale = pgs[((size_t)b * TT + t) * 2 + 1];
    const float inv = scale / rs[0];
    float* orow = pfin + (size_t)r * VV;
    for (int v = tid; v < VV; v += 256)
        orow[v] = __expf(lrow[v] - M) * inv;
}

__global__ void scatter_kernel(const int* __restrict__ index_list, const float* __restrict__ wts,
                               const float* __restrict__ pgs, const int* __restrict__ sotp,
                               float* __restrict__ pfin)
{
    const int sot  = *sotp;
    const int tlen = TT - sot;
    const int nrows = BB * tlen;
    const int gidx = blockIdx.x * blockDim.x + threadIdx.x;
    const int r = gidx >> 7, m = gidx & (MM - 1);
    if (r >= nrows) return;
    const int b = r / tlen, tt = r % tlen, t = tt + sot - 1;
    const int bt = b * TT + t;
    const int id = index_list[(size_t)bt * MM + m];
    if (id < VV) {
        float val = pgs[bt * 2] * wts[(size_t)bt * MM + m];
        atomicAdd(pfin + (size_t)r * VV + id, val);
    }
}

__global__ void init_accum_kernel(float* __restrict__ accum)
{
    if (threadIdx.x < 2) accum[threadIdx.x] = 0.f;
}

__global__ void loss_kernel(const float* __restrict__ pfin, const int* __restrict__ targets,
                            const int* __restrict__ sotp, float* __restrict__ accum)
{
    const int sot  = *sotp;
    const int tlen = TT - sot;
    const int nrows = BB * tlen;
    const int r = blockIdx.x * blockDim.x + threadIdx.x;
    if (r >= nrows) return;
    const int b = r / tlen, tt = r % tlen;
    const int tg = targets[b * TT + tt + sot];
    float pt = 0.f;
    if (tg != -100) pt = -logf(pfin[(size_t)r * VV + tg] + 1e-9f);
    atomicAdd(accum, pt);
    atomicAdd(accum + 1, (pt != 0.f) ? 1.f : 0.f);
}

__global__ void finalize_kernel(const float* __restrict__ accum, float* __restrict__ out)
{
    if (threadIdx.x == 0 && blockIdx.x == 0) out[0] = accum[0] / accum[1];
}

extern "C" void kernel_launch(void* const* d_in, const int* in_sizes, int n_in,
                              void* d_out, int out_size, void* d_ws, size_t ws_size,
                              hipStream_t stream)
{
    const float* hidden     = (const float*)d_in[0];
    const float* logits     = (const float*)d_in[1];
    const float* tok_emb    = (const float*)d_in[2];
    const float* ooKB_emb   = (const float*)d_in[3];
    const float* Wq         = (const float*)d_in[4];
    const float* bq         = (const float*)d_in[5];
    const float* Wk         = (const float*)d_in[6];
    const float* bk         = (const float*)d_in[7];
    const float* Wg         = (const float*)d_in[8];
    const float* bg         = (const float*)d_in[9];
    const int*   targets    = (const int*)d_in[10];
    const int*   index_list = (const int*)d_in[11];
    const int*   step_mask  = (const int*)d_in[12];
    const int*   p_gen_mask = (const int*)d_in[13];
    const int*   sotp       = (const int*)d_in[14];

    float* ws     = (float*)d_ws;
    float* KBproj = ws;                                      // (V+1) * 512
    float* query  = KBproj + (size_t)(VV + 1) * AA;          // B*T * 512
    float* wts    = query  + (size_t)BB * TT * AA;           // B*T * 128
    float* pgs    = wts    + (size_t)BB * TT * MM;           // B*T * 2  (pg, scale)
    float* accum  = pgs    + (size_t)BB * TT * 2;            // 2

    float* out  = (float*)d_out;
    float* pfin = out + 1;                                   // p_final, row-major (nrows x V)

    // 1) KBproj = concat(tok_emb, ooKB) @ Wk^T + bk    (WMMA bf16, f32 accumulate)
    gemm_xwT_kernel<<<dim3((VV + 1 + 63) / 64, 8), dim3(128), 0, stream>>>(
        tok_emb, ooKB_emb, VV + 1, VV, Wk, bk, KBproj);

    // 2) query = hidden @ Wq^T + bq
    gemm_xwT_kernel<<<dim3((BB * TT + 63) / 64, 8), dim3(128), 0, stream>>>(
        hidden, nullptr, BB * TT, BB * TT, Wq, bq, query);

    // 3) attention weights, pg, scale per (b,t)
    attend_kernel<<<dim3(BB * TT), dim3(256), 0, stream>>>(
        KBproj, query, hidden, index_list, step_mask, p_gen_mask, Wg, bg, wts, pgs);

    // 4) zero loss accumulators
    init_accum_kernel<<<1, 32, 0, stream>>>(accum);

    // 5) p_final = softmax(logits) * (1 - pg + comp)
    row_softmax_kernel<<<dim3(BB * TT), dim3(256), 0, stream>>>(logits, pgs, sotp, pfin);

    // 6) scatter pg * w into p_final
    scatter_kernel<<<dim3((BB * TT * MM) / 256), dim3(256), 0, stream>>>(
        index_list, wts, pgs, sotp, pfin);

    // 7) per-token NLL, 8) loss = sum / count
    loss_kernel<<<dim3((BB * TT + 255) / 256), dim3(256), 0, stream>>>(pfin, targets, sotp, accum);
    finalize_kernel<<<1, 32, 0, stream>>>(accum, out);
}
